// Lenet5v5_7971459301395
// MI455X (gfx1250) — 8-bit matrix operand order repaired and checked by reading, compiled, not run on hardware
//
#include <hip/hip_runtime.h>
#include <cstdint>
#include <cstddef>

typedef __attribute__((ext_vector_type(8))) int v8i;

#define EPS_BN 1e-5f
#define EPS_TN 1e-4f

__device__ __forceinline__ float qf(float x) {
  return fminf(fmaxf(rintf(x), -1.0f), 1.0f);
}

__global__ void pack_w_kernel(const float* __restrict__ src, int8_t* __restrict__ dst,
                              int N, int K, int Npad, int Kpad) {
  int t = blockIdx.x * blockDim.x + threadIdx.x;
  if (t >= Npad * Kpad) return;
  int n = t / Kpad, k = t - n * Kpad;
  float v = 0.0f;
  if (n < N && k < K) v = qf(src[(size_t)n * K + k]);
  dst[t] = (int8_t)__float2int_rn(v);
}

__global__ void pack_wc1_kernel(const float* __restrict__ src, int8_t* __restrict__ dst) {
  int t = blockIdx.x * blockDim.x + threadIdx.x;
  if (t >= 32 * 128) return;
  int n = t >> 7, k = t & 127;
  int g = k >> 3, kx = k & 7;
  float v = 0.0f;
  if (n < 18 && g < 15 && kx < 5) v = qf(src[n * 75 + g * 5 + kx]);
  dst[t] = (int8_t)__float2int_rn(v);
}

__global__ void quant_in_kernel(const float* __restrict__ x, int8_t* __restrict__ xq, int n) {
  int t = blockIdx.x * blockDim.x + threadIdx.x;
  if (t >= n) return;
  xq[t] = (int8_t)__float2int_rn(qf(2.0f * x[t] - 1.0f));
}

__global__ void zero16_kernel(int4* __restrict__ p, int n16) {
  int t = blockIdx.x * blockDim.x + threadIdx.x;
  if (t < n16) p[t] = make_int4(0, 0, 0, 0);
}

__global__ void im2col1_kernel(const int8_t* __restrict__ xq, int8_t* __restrict__ col, int b0) {
  int t = blockIdx.x * blockDim.x + threadIdx.x;
  if (t >= 200704 * 15) return;
  int g    = t % 15;
  int mloc = t / 15;
  int ox = mloc % 28;
  int oy = (mloc / 28) % 28;
  int b  = b0 + mloc / 784;
  int ci = g / 5, ky = g - ci * 5;
  const int8_t* s = xq + ((size_t)(b * 3 + ci) * 32 + (oy + ky)) * 32 + ox;
  int lo = ((int)s[0] & 0xFF) | (((int)s[1] & 0xFF) << 8) | (((int)s[2] & 0xFF) << 16) | (((int)s[3] & 0xFF) << 24);
  int hi = ((int)s[4] & 0xFF) | (((int)s[5] & 0xFF) << 8) | (((int)s[6] & 0xFF) << 16) | (((int)s[7] & 0xFF) << 24);
  *(int2*)(col + (size_t)mloc * 128 + g * 8) = make_int2(lo, hi);
}

__global__ void im2col2_kernel(const int8_t* __restrict__ a1, int8_t* __restrict__ col, int b0) {
  int t = blockIdx.x * blockDim.x + threadIdx.x;
  if (t >= 51200 * 128) return;
  int dw   = t & 127;
  int mloc = t >> 7;
  int ox = mloc % 10;
  int oy = (mloc / 10) % 10;
  int b  = b0 + mloc / 100;
  int d = 0;
#pragma unroll
  for (int i = 0; i < 4; ++i) {
    int k  = dw * 4 + i;
    int kc = (k < 450) ? k : 0;
    int ci = kc / 25;
    int r  = kc - ci * 25;
    int ky = r / 5;
    int kx = r - ky * 5;
    int v = (int)a1[((size_t)(b * 18 + ci) * 14 + (oy + ky)) * 14 + (ox + kx)] & 0xFF;
    v = (k < 450) ? v : 0;
    d |= v << (8 * i);
  }
  *(int*)(col + (size_t)mloc * 512 + dw * 4) = d;
}

template <int MODE>
__global__ void gemm_iu8_kernel(const int8_t* __restrict__ A, const int8_t* __restrict__ B,
                                int Kpad, int Ntiles, int TotalTiles,
                                const float* __restrict__ gg, const float* __restrict__ bb,
                                const float* __restrict__ mm, const float* __restrict__ vv,
                                int Nact, int ldout,
                                int8_t* __restrict__ out8, float* __restrict__ outf,
                                const float* __restrict__ tnw, const float* __restrict__ tnb,
                                const float* __restrict__ tnm, const float* __restrict__ tnv) {
  const int lane = threadIdx.x & 31;
  const int wid  = threadIdx.x >> 5;
  const int tile = blockIdx.x * 8 + wid;
  if (tile >= TotalTiles) return;
  const int mt   = tile / Ntiles;
  const int nt   = tile - mt * Ntiles;
  const int half = lane >> 4;
  const int l16  = lane & 15;

  const int8_t* arow = A + (size_t)(mt * 16 + l16) * Kpad;
  const int8_t* brow = B + (size_t)(nt * 16 + l16) * Kpad;

  v8i acc = {0, 0, 0, 0, 0, 0, 0, 0};
  for (int k0 = 0; k0 < Kpad; k0 += 64) {
    int2 a01 = *(const int2*)(arow + k0 + half * 8);
    int2 a23 = *(const int2*)(arow + k0 + half * 8 + 16);
    int2 a45 = *(const int2*)(arow + k0 + half * 8 + 32);
    int2 a67 = *(const int2*)(arow + k0 + half * 8 + 48);
    int2 b01 = *(const int2*)(brow + k0 + half * 8);
    int2 b23 = *(const int2*)(brow + k0 + half * 8 + 16);
    int2 b45 = *(const int2*)(brow + k0 + half * 8 + 32);
    int2 b67 = *(const int2*)(brow + k0 + half * 8 + 48);
    int4 bA = make_int4(b01.x, b01.y, b23.x, b23.y);
    int4 bB = make_int4(b45.x, b45.y, b67.x, b67.y);
    v8i af, bfr;
    af[0] = a01.x; af[1] = a01.y; af[2] = a23.x; af[3] = a23.y;
    af[4] = a45.x; af[5] = a45.y; af[6] = a67.x; af[7] = a67.y;
    bfr[0] = bA.x; bfr[1] = bA.y; bfr[2] = bA.z; bfr[3] = bA.w;
    bfr[4] = bB.x; bfr[5] = bB.y; bfr[6] = bB.z; bfr[7] = bB.w;
    acc = __builtin_amdgcn_wmma_i32_16x16x64_iu8(true, af, true, bfr, acc, false, false);
  }

  const int n = nt * 16 + l16;
  const bool valid_n = (n < Nact);
  float sc = 0.0f, mu = 0.0f, be = 0.0f;
  if (valid_n) {
    sc = gg[n] / sqrtf(vv[n] + EPS_BN);
    mu = mm[n];
    be = bb[n];
  }
  float tscale = 0.0f, tbias = 0.0f, tmean = 0.0f;
  if (MODE == 2) {
    tscale = tnw[0] / sqrtf(tnv[0] + EPS_TN);
    tbias  = tnb[0];
    tmean  = tnm[0];
  }
#pragma unroll
  for (int r = 0; r < 8; ++r) {
    int m = mt * 16 + r + half * 8;
    float f = fmaxf((float)acc[r], 0.0f);
    f = (f - mu) * sc + be;
    float qv = qf(f);
    if (MODE == 1) {
      if (valid_n) out8[(size_t)m * ldout + n] = (int8_t)__float2int_rn(qv);
    } else {
      if (valid_n) outf[(size_t)m * 10 + n] = (qv - tmean) * tscale + tbias;
    }
  }
}

__global__ void pool1_kernel(const int8_t* __restrict__ c1t, int8_t* __restrict__ a1, int b0) {
  int t = blockIdx.x * blockDim.x + threadIdx.x;
  if (t >= 256 * 18 * 196) return;
  int px = t % 14;
  int py = (t / 14) % 14;
  int c  = (t / 196) % 18;
  int bl = t / 3528;
  int v = -2;
#pragma unroll
  for (int dy = 0; dy < 2; ++dy)
#pragma unroll
    for (int dx = 0; dx < 2; ++dx) {
      int mloc = bl * 784 + (2 * py + dy) * 28 + (2 * px + dx);
      v = max(v, (int)c1t[(size_t)mloc * 18 + c]);
    }
  a1[((size_t)((b0 + bl) * 18 + c) * 14 + py) * 14 + px] = (int8_t)v;
}

__global__ void pool2_kernel(const int8_t* __restrict__ c2t, int8_t* __restrict__ a2, int b0) {
  int t = blockIdx.x * blockDim.x + threadIdx.x;
  if (t >= 512 * 48 * 25) return;
  int px = t % 5;
  int py = (t / 5) % 5;
  int c  = (t / 25) % 48;
  int bl = t / 1200;
  int v = -2;
#pragma unroll
  for (int dy = 0; dy < 2; ++dy)
#pragma unroll
    for (int dx = 0; dx < 2; ++dx) {
      int mloc = bl * 100 + (2 * py + dy) * 10 + (2 * px + dx);
      v = max(v, (int)c2t[(size_t)mloc * 48 + c]);
    }
  a2[(size_t)(b0 + bl) * 1216 + c * 25 + py * 5 + px] = (int8_t)v;
}

extern "C" void kernel_launch(void* const* d_in, const int* in_sizes, int n_in,
                              void* d_out, int out_size, void* d_ws, size_t ws_size,
                              hipStream_t stream) {
  (void)in_sizes; (void)n_in; (void)out_size; (void)ws_size;

  const float* x   = (const float*)d_in[0];
  const float* wc1 = (const float*)d_in[1];
  const float* wc2 = (const float*)d_in[2];
  const float* g1 = (const float*)d_in[3],  *b1 = (const float*)d_in[4];
  const float* m1 = (const float*)d_in[5],  *v1 = (const float*)d_in[6];
  const float* g2 = (const float*)d_in[7],  *b2 = (const float*)d_in[8];
  const float* m2 = (const float*)d_in[9],  *v2 = (const float*)d_in[10];
  const float* g3 = (const float*)d_in[11], *b3 = (const float*)d_in[12];
  const float* m3 = (const float*)d_in[13], *v3 = (const float*)d_in[14];
  const float* g4 = (const float*)d_in[15], *b4 = (const float*)d_in[16];
  const float* m4 = (const float*)d_in[17], *v4 = (const float*)d_in[18];
  const float* g5 = (const float*)d_in[19], *b5 = (const float*)d_in[20];
  const float* m5 = (const float*)d_in[21], *v5 = (const float*)d_in[22];
  const float* wf1 = (const float*)d_in[23];
  const float* wf2 = (const float*)d_in[24];
  const float* wf3 = (const float*)d_in[25];
  const float* tnw = (const float*)d_in[26], *tnb = (const float*)d_in[27];
  const float* tnm = (const float*)d_in[28], *tnv = (const float*)d_in[29];
  float* out = (float*)d_out;

  char* base = (char*)d_ws;
  size_t off = 0;
  auto carve = [&](size_t bytes) -> char* {
    char* p = base + off;
    off += (bytes + 255) & ~(size_t)255;
    return p;
  };
  int8_t* wq1  = (int8_t*)carve((size_t)32 * 128);
  int8_t* wq2  = (int8_t*)carve((size_t)48 * 512);
  int8_t* wqf1 = (int8_t*)carve((size_t)368 * 1216);
  int8_t* wqf2 = (int8_t*)carve((size_t)256 * 384);
  int8_t* wqf3 = (int8_t*)carve((size_t)16 * 256);
  int8_t* xq   = (int8_t*)carve((size_t)4096 * 3072 + 256);
  int8_t* a1   = (int8_t*)carve((size_t)4096 * 18 * 196 + 256);
  int8_t* col  = (int8_t*)carve((size_t)51200 * 512);
  int8_t* c1t  = (int8_t*)carve((size_t)200704 * 18);
  int8_t* c2t  = (int8_t*)carve((size_t)51200 * 48);
  int8_t* a2   = (int8_t*)carve((size_t)4096 * 1216);
  int8_t* a3   = (int8_t*)carve((size_t)4096 * 384);
  int8_t* a4   = (int8_t*)carve((size_t)4096 * 256);

  const int T = 256;
  auto blks = [](long n, int t) { return (unsigned)((n + t - 1) / t); };

  pack_wc1_kernel<<<blks(32 * 128, T), T, 0, stream>>>(wc1, wq1);
  pack_w_kernel<<<blks(48 * 512, T), T, 0, stream>>>(wc2, wq2, 48, 450, 48, 512);
  pack_w_kernel<<<blks((long)368 * 1216, T), T, 0, stream>>>(wf1, wqf1, 360, 1200, 368, 1216);
  pack_w_kernel<<<blks((long)256 * 384, T), T, 0, stream>>>(wf2, wqf2, 252, 360, 256, 384);
  pack_w_kernel<<<blks((long)16 * 256, T), T, 0, stream>>>(wf3, wqf3, 10, 252, 16, 256);

  quant_in_kernel<<<blks((long)4096 * 3072, T), T, 0, stream>>>(x, xq, 4096 * 3072);

  zero16_kernel<<<blks(7602176 / 16, T), T, 0, stream>>>((int4*)a2, 7602176 / 16);

  for (int ch = 0; ch < 16; ++ch) {
    int b0 = ch * 256;
    im2col1_kernel<<<blks((long)200704 * 15, T), T, 0, stream>>>(xq, col, b0);
    gemm_iu8_kernel<1><<<(12544 * 2) / 8, T, 0, stream>>>(col, wq1, 128, 2, 12544 * 2,
                                                          g1, b1, m1, v1, 18, 18, c1t, nullptr,
                                                          tnw, tnb, tnm, tnv);
    pool1_kernel<<<blks((long)256 * 3528, T), T, 0, stream>>>(c1t, a1, b0);
  }

  for (int ch = 0; ch < 8; ++ch) {
    int b0 = ch * 512;
    im2col2_kernel<<<blks((long)51200 * 128, T), T, 0, stream>>>(a1, col, b0);
    gemm_iu8_kernel<1><<<(3200 * 3) / 8, T, 0, stream>>>(col, wq2, 512, 3, 3200 * 3,
                                                         g2, b2, m2, v2, 48, 48, c2t, nullptr,
                                                         tnw, tnb, tnm, tnv);
    pool2_kernel<<<blks((long)512 * 1200, T), T, 0, stream>>>(c2t, a2, b0);
  }

  gemm_iu8_kernel<1><<<(256 * 23) / 8, T, 0, stream>>>(a2, wqf1, 1216, 23, 256 * 23,
                                                       g3, b3, m3, v3, 360, 384, a3, nullptr,
                                                       tnw, tnb, tnm, tnv);
  gemm_iu8_kernel<1><<<(256 * 16) / 8, T, 0, stream>>>(a3, wqf2, 384, 16, 256 * 16,
                                                       g4, b4, m4, v4, 252, 256, a4, nullptr,
                                                       tnw, tnb, tnm, tnv);
  gemm_iu8_kernel<2><<<256 / 8, T, 0, stream>>>(a4, wqf3, 256, 1, 256,
                                                g5, b5, m5, v5, 10, 0, nullptr, out,
                                                tnw, tnb, tnm, tnv);
}
